// RadarPillars_49340584296778
// MI455X (gfx1250) — matrix operand order repaired and checked by reading, not compiled, not run on hardware
//
#include <hip/hip_runtime.h>
#include <math.h>

// ---------------- types / helpers ----------------
typedef __attribute__((ext_vector_type(16))) __bf16          v16bf;
typedef __attribute__((ext_vector_type(16))) unsigned short  v16u;
typedef __attribute__((ext_vector_type(8)))  unsigned short  v8u;
typedef __attribute__((ext_vector_type(8)))  float           v8f;

#define DEVINL static __device__ __forceinline__

DEVINL unsigned short f2bf(float f) {
    unsigned int u = __builtin_bit_cast(unsigned int, f);
    u += 0x7FFFu + ((u >> 16) & 1u);
    return (unsigned short)(u >> 16);
}

DEVINL v8f wmma_bf16(v16u a, v16u b, v8f c) {
    return __builtin_amdgcn_wmma_f32_16x16x32_bf16(
        false, __builtin_bit_cast(v16bf, a),
        false, __builtin_bit_cast(v16bf, b),
        (short)0, c, false, false);
}

// ---- cross-lane reductions: DPP row_ror tree (VALU only, no LDS) ----
template <int CTRL>
DEVINL float dppf(float v) {
    int x = __builtin_bit_cast(int, v);
    int r = __builtin_amdgcn_update_dpp(x, x, CTRL, 0xF, 0xF, true);
    return __builtin_bit_cast(float, r);
}
// reduction across a 16-lane row (matches WMMA C/D half-wave row groups)
DEVINL float red16_max(float v) {
    v = fmaxf(v, dppf<0x128>(v));  // row_ror:8
    v = fmaxf(v, dppf<0x124>(v));  // row_ror:4
    v = fmaxf(v, dppf<0x122>(v));  // row_ror:2
    v = fmaxf(v, dppf<0x121>(v));  // row_ror:1
    return v;
}
DEVINL float red16_sum(float v) {
    v += dppf<0x128>(v);
    v += dppf<0x124>(v);
    v += dppf<0x122>(v);
    v += dppf<0x121>(v);
    return v;
}
// full-wave (32 lane) reductions: DPP tree + one cross-half shuffle
DEVINL float red32_sum(float v) { v = red16_sum(v); v += __shfl_xor(v, 16); return v; }
DEVINL float red32_max(float v) { v = red16_max(v); v = fmaxf(v, __shfl_xor(v, 16)); return v; }

// A-matrix per-lane layout (16x32 bf16): two contiguous 8-elem channel segments
DEVINL v16u loadA_f32(const float* rowp, int s0, int s1) {
    v16u r;
#pragma unroll
    for (int t = 0; t < 8; ++t) r[t] = f2bf(rowp[s0 + t]);
#pragma unroll
    for (int t = 0; t < 8; ++t) r[8 + t] = f2bf(rowp[s1 + t]);
    return r;
}
DEVINL v16u loadA_bf(const unsigned short* rowp, int s0, int s1) {
    v8u a = *(const v8u*)(rowp + s0);
    v8u b = *(const v8u*)(rowp + s1);
    v16u r;
#pragma unroll
    for (int t = 0; t < 8; ++t) { r[t] = a[t]; r[8 + t] = b[t]; }
    return r;
}
// B-matrix per-lane layout: 16 contiguous K elements of (transposed) operand row
DEVINL v16u loadB_bf(const unsigned short* p) { return *(const v16u*)p; }

// ---------------- small utility kernels ----------------
__global__ void zero_kernel(float* p, int n) {
    int i = blockIdx.x * blockDim.x + threadIdx.x;
    if (i < n) p[i] = 0.0f;
}

// transpose weights to bf16:  wT[n*K+k] = w[k*N+n]
__global__ void prepw_kernel(const float* wq, const float* wk, const float* wv,
                             const float* wo, const float* wf1, const float* wf2,
                             unsigned short* wqT, unsigned short* wkT,
                             unsigned short* wvT, unsigned short* woT,
                             unsigned short* wf1T, unsigned short* wf2T) {
    int i = blockIdx.x * blockDim.x + threadIdx.x;
    if (i < 4096) {
        int n = i >> 6, k = i & 63;
        wqT[i] = f2bf(wq[k * 64 + n]);
        wkT[i] = f2bf(wk[k * 64 + n]);
        wvT[i] = f2bf(wv[k * 64 + n]);
        woT[i] = f2bf(wo[k * 64 + n]);
    }
    if (i < 16384) {
        { int n = i >> 6, k = i & 63;   wf1T[i] = f2bf(wf1[k * 256 + n]); }  // (64,256)->(256,64)
        { int n = i >> 8, k = i & 255;  wf2T[i] = f2bf(wf2[k * 64 + n]);  }  // (256,64)->(64,256)
    }
}

// ---------------- PFN (3-pass, recompute, global BN) ----------------
#define PFN_N 262144  // B*P*NPTS

DEVINL void compute_h1(const float* pts, const float* cc, int gp, int lane,
                       const float* w1, const float* b1, float* h1) {
    const float* pp = pts + (((size_t)gp * 32 + lane) * 4);
    const float* c  = cc + (size_t)gp * 3;
    float f0 = pp[0], f1 = pp[1], f2 = pp[2], f3 = pp[3];
    float f4 = f0 - c[0], f5 = f1 - c[1], f6 = f2 - c[2];
#pragma unroll
    for (int o = 0; o < 64; ++o)
        h1[o] = b1[o] + f0 * w1[o] + f1 * w1[64 + o] + f2 * w1[128 + o] +
                f3 * w1[192 + o] + f4 * w1[256 + o] + f5 * w1[320 + o] +
                f6 * w1[384 + o];
}

DEVINL void norm_h1(float* h1, const float* bn, const float* g1, const float* be1) {
    const float invN = 1.0f / PFN_N;
#pragma unroll
    for (int o = 0; o < 64; ++o) {
        float mu = bn[o] * invN;
        float var = bn[64 + o] * invN - mu * mu;
        h1[o] = fmaxf((h1[o] - mu) * rsqrtf(var + 1e-5f) * g1[o] + be1[o], 0.0f);
    }
}

__global__ __launch_bounds__(256, 1) void pfn_pass1(const float* pts,
                                                    const float* cc,
                                                    const float* w1,
                                                    const float* b1, float* bn) {
    int lane = threadIdx.x & 31, wid = threadIdx.x >> 5;
    int gp = blockIdx.x * 8 + wid;
    float h1[64];
    compute_h1(pts, cc, gp, lane, w1, b1, h1);
#pragma unroll
    for (int o = 0; o < 64; ++o) {
        float s = red32_sum(h1[o]);
        float q = red32_sum(h1[o] * h1[o]);
        if (lane == 0) { atomicAdd(&bn[o], s); atomicAdd(&bn[64 + o], q); }
    }
}

__global__ __launch_bounds__(256, 1) void pfn_pass2(
    const float* pts, const float* cc, const float* w1, const float* b1,
    const float* g1, const float* be1, const float* w2, const float* b2,
    float* bn) {
    int lane = threadIdx.x & 31, wid = threadIdx.x >> 5;
    int gp = blockIdx.x * 8 + wid;
    float h1[64];
    compute_h1(pts, cc, gp, lane, w1, b1, h1);
    norm_h1(h1, bn, g1, be1);
    for (int o = 0; o < 64; ++o) {
        float a = b2[o];
#pragma unroll
        for (int i = 0; i < 64; ++i) a += h1[i] * w2[i * 64 + o];
        float s = red32_sum(a);
        float q = red32_sum(a * a);
        if (lane == 0) { atomicAdd(&bn[128 + o], s); atomicAdd(&bn[192 + o], q); }
    }
}

__global__ __launch_bounds__(256, 1) void pfn_pass3(
    const float* pts, const float* cc, const float* w1, const float* b1,
    const float* g1, const float* be1, const float* w2, const float* b2,
    const float* g2, const float* be2, const float* bn, float* pillar) {
    int lane = threadIdx.x & 31, wid = threadIdx.x >> 5;
    int gp = blockIdx.x * 8 + wid;
    float h1[64];
    compute_h1(pts, cc, gp, lane, w1, b1, h1);
    norm_h1(h1, bn, g1, be1);
    const float invN = 1.0f / PFN_N;
    for (int o = 0; o < 64; ++o) {
        float a = b2[o];
#pragma unroll
        for (int i = 0; i < 64; ++i) a += h1[i] * w2[i * 64 + o];
        float mu = bn[128 + o] * invN;
        float var = bn[192 + o] * invN - mu * mu;
        a = fmaxf((a - mu) * rsqrtf(var + 1e-5f) * g2[o] + be2[o], 0.0f);
        a = red32_max(a);
        if (lane == 0) pillar[(size_t)gp * 64 + o] = a;
    }
}

// ---------------- scatter to BEV grid (B,H,W,C) ----------------
__global__ void scatter_kernel(const float* pillar, const float* cc, float* grid) {
    int pb = blockIdx.x;          // b*P + p
    int c = threadIdx.x;          // channel 0..63
    float cx = cc[pb * 3 + 0], cy = cc[pb * 3 + 1];
    int xi = (int)cx; xi = xi < 0 ? 0 : (xi > 63 ? 63 : xi);
    int yi = (int)cy; yi = yi < 0 ? 0 : (yi > 63 ? 63 : yi);
    int b = pb >> 11;
    grid[(((size_t)(b * 64 + yi) * 64 + xi)) * 64 + c] = pillar[(size_t)pb * 64 + c];
}

// ---------------- LayerNorm over C=64, one wave per row ----------------
__global__ void ln_kernel(const float* __restrict__ x, const float* g,
                          const float* be, float* __restrict__ out, int rows) {
    int wid = (blockIdx.x * blockDim.x + threadIdx.x) >> 5;
    int lane = threadIdx.x & 31;
    if (wid >= rows) return;
    const float* r = x + (size_t)wid * 64;
    float a = r[lane], b = r[lane + 32];
    float mu = red32_sum(a + b) * (1.0f / 64.0f);
    float da = a - mu, db = b - mu;
    float var = red32_sum(da * da + db * db) * (1.0f / 64.0f);
    float inv = rsqrtf(var + 1e-5f);
    out[(size_t)wid * 64 + lane]      = da * inv * g[lane] + be[lane];
    out[(size_t)wid * 64 + lane + 32] = db * inv * g[lane + 32] + be[lane + 32];
}

// ---------------- generic WMMA GEMM:  out = f(A @ BT^T + bias) ----------------
// out_mode: 0/3 f32 row-major (3: += resid), 1 bf16 row-major, 2 bf16 transposed per batch (seq=4096)
// act: 0 none, 1 exact gelu
template <int ABF>
__global__ __launch_bounds__(256, 1) void gemm_wmma_kernel(
    const void* __restrict__ Ap, const unsigned short* __restrict__ BT,
    const float* __restrict__ bias, int M, int N, int K, void* __restrict__ outp,
    int out_mode, float scale, const float* __restrict__ resid, int act) {
    int lane = threadIdx.x & 31;
    int wid = threadIdx.x >> 5;
    int tile = blockIdx.x * (blockDim.x >> 5) + wid;
    int tn = N >> 4;
    int tm_i = tile / tn, tn_i = tile % tn;
    if (tm_i >= (M >> 4)) return;
    int half = lane >> 4, l16 = lane & 15;
    int arow = tm_i * 16 + l16;
    int brow = tn_i * 16 + l16;
    const float* Af = (const float*)Ap;
    const unsigned short* Ab = (const unsigned short*)Ap;
    v8f acc = {0, 0, 0, 0, 0, 0, 0, 0};
    for (int k0 = 0; k0 < K; k0 += 32) {
        int s0 = k0 + half * 16, s1 = k0 + half * 16 + 8;
        v16u a = ABF ? loadA_bf(Ab + (size_t)arow * K, s0, s1)
                     : loadA_f32(Af + (size_t)arow * K, s0, s1);
        v16u b = loadB_bf(BT + (size_t)brow * K + k0 + half * 16);
        acc = wmma_bf16(a, b, acc);
    }
    int coln = tn_i * 16 + l16;
    float bsv = bias[coln];
#pragma unroll
    for (int r = 0; r < 8; ++r) {
        int row = tm_i * 16 + r + half * 8;
        float v = acc[r] + bsv;
        if (act == 1) v = 0.5f * v * (1.0f + erff(v * 0.70710678118f));
        v *= scale;
        if (resid) v += resid[(size_t)row * N + coln];
        if (out_mode == 1) {
            ((unsigned short*)outp)[(size_t)row * N + coln] = f2bf(v);
        } else if (out_mode == 2) {
            int b = row >> 12, s = row & 4095;
            ((unsigned short*)outp)[((size_t)(b * N + coln) << 12) + s] = f2bf(v);
        } else {
            ((float*)outp)[(size_t)row * N + coln] = v;
        }
    }
}

// ---------------- flash attention: wave = 16-query tile, 32-key chunks ----------------
__global__ __launch_bounds__(128, 1) void flash_attn(
    const unsigned short* __restrict__ qb, const unsigned short* __restrict__ kb,
    const unsigned short* __restrict__ vtb, const int* __restrict__ mask,
    float* __restrict__ o) {
    __shared__ __align__(64) unsigned short sP[4][16][32];
    int lane = threadIdx.x & 31, wid = threadIdx.x >> 5;
    int gw = blockIdx.x * 4 + wid;   // 0..1023
    int b = gw >> 8;
    int qt = gw & 255;
    int half = lane >> 4, l16 = lane & 15;

    const unsigned short* qrow = qb + ((size_t)(b * 4096 + qt * 16 + l16)) * 64;
    v16u aq0 = loadA_bf(qrow, half * 16, half * 16 + 8);        // channels 0..31
    v16u aq1 = loadA_bf(qrow, 32 + half * 16, 40 + half * 16);   // channels 32..63

    v8f O[4];
#pragma unroll
    for (int cg = 0; cg < 4; ++cg) O[cg] = (v8f){0, 0, 0, 0, 0, 0, 0, 0};
    float mrow[8], lrow[8];
#pragma unroll
    for (int r = 0; r < 8; ++r) { mrow[r] = -1e30f; lrow[r] = 0.0f; }

    const unsigned short* kbase = kb + ((size_t)b << 12) * 64;
    const unsigned short* vbase = vtb + ((size_t)(b * 64) << 12);
    const int* mb = mask + (b << 12);

    for (int kt = 0; kt < 4096; kt += 32) {
        v8f s0 = {0, 0, 0, 0, 0, 0, 0, 0}, s1 = {0, 0, 0, 0, 0, 0, 0, 0};
        {
            const unsigned short* kr0 = kbase + (size_t)(kt + l16) * 64;
            const unsigned short* kr1 = kbase + (size_t)(kt + 16 + l16) * 64;
            s0 = wmma_bf16(aq0, loadB_bf(kr0 + half * 16), s0);
            s0 = wmma_bf16(aq1, loadB_bf(kr0 + 32 + half * 16), s0);
            s1 = wmma_bf16(aq0, loadB_bf(kr1 + half * 16), s1);
            s1 = wmma_bf16(aq1, loadB_bf(kr1 + 32 + half * 16), s1);
        }
        int mk0 = mb[kt + l16], mk1 = mb[kt + 16 + l16];
        if (!mk0) {
#pragma unroll
            for (int r = 0; r < 8; ++r) s0[r] = -1e9f;
        }
        if (!mk1) {
#pragma unroll
            for (int r = 0; r < 8; ++r) s1[r] = -1e9f;
        }
        float fac[8];
#pragma unroll
        for (int r = 0; r < 8; ++r) {
            float mx = red16_max(fmaxf(s0[r], s1[r]));   // DPP row reduction, no LDS
            float mn = fmaxf(mrow[r], mx);
            float f = __expf(mrow[r] - mn);
            float p0 = __expf(s0[r] - mn), p1 = __expf(s1[r] - mn);
            float ps = red16_sum(p0 + p1);               // DPP row reduction, no LDS
            lrow[r] = lrow[r] * f + ps;
            mrow[r] = mn;
            fac[r] = f;
            int rr = r + half * 8;
            sP[wid][rr][l16] = f2bf(p0);
            sP[wid][rr][16 + l16] = f2bf(p1);
        }
#pragma unroll
        for (int cg = 0; cg < 4; ++cg)
#pragma unroll
            for (int r = 0; r < 8; ++r) O[cg][r] *= fac[r];
        asm volatile("s_wait_dscnt 0" ::: "memory");  // intra-wave LDS RAW (P staging)
        v16u ap = loadA_bf(&sP[wid][l16][0], half * 16, half * 16 + 8);
#pragma unroll
        for (int cg = 0; cg < 4; ++cg) {
            const unsigned short* vr =
                vbase + (((size_t)(cg * 16 + l16)) << 12) + kt + half * 16;
            O[cg] = wmma_bf16(ap, loadB_bf(vr), O[cg]);
        }
    }
    float* ob = o + ((size_t)(b * 4096 + qt * 16)) * 64;
#pragma unroll
    for (int r = 0; r < 8; ++r) {
        float inv = 1.0f / lrow[r];
        int row = r + half * 8;
#pragma unroll
        for (int cg = 0; cg < 4; ++cg)
            ob[(size_t)row * 64 + cg * 16 + l16] = O[cg][r] * inv;
    }
}

// ---------------- build x = pseudo + masked attended (NCHW) ----------------
__global__ void buildx_kernel(const float* grid, const float* hh2,
                              const int* mask, float* x) {
    int idx = blockIdx.x * blockDim.x + threadIdx.x;
    if (idx >= 4 * 64 * 64 * 64) return;
    int xc = idx & 63;
    int y = (idx >> 6) & 63;
    int c = (idx >> 12) & 63;
    int b = idx >> 18;
    int s = y * 64 + xc;
    float att = mask[b * 4096 + s] ? hh2[((size_t)(b * 4096 + s)) * 64 + c] : 0.0f;
    x[idx] = grid[((size_t)(b * 4096 + s)) * 64 + c] + att;
}

// ---------------- convs ----------------
__global__ void conv3x3_kernel(const float* __restrict__ in,
                               const float* __restrict__ w,
                               const float* __restrict__ bias,
                               float* __restrict__ out, int Bn, int Cin, int Cout,
                               int Hi, int Wi, int Ho, int Wo, int stride,
                               int relu) {
    int idx = blockIdx.x * blockDim.x + threadIdx.x;
    int total = Bn * Cout * Ho * Wo;
    if (idx >= total) return;
    int x = idx % Wo;
    int y = (idx / Wo) % Ho;
    int co = (idx / (Wo * Ho)) % Cout;
    int b = idx / (Wo * Ho * Cout);
    float acc = bias[co];
    int iy0 = y * stride - 1, ix0 = x * stride - 1;
    int HiWi = Hi * Wi;
    for (int ky = 0; ky < 3; ++ky) {
        int iy = iy0 + ky;
        if ((unsigned)iy >= (unsigned)Hi) continue;
        for (int kx = 0; kx < 3; ++kx) {
            int ix = ix0 + kx;
            if ((unsigned)ix >= (unsigned)Wi) continue;
            const float* ip = in + (((size_t)(b * Cin) * Hi + iy) * Wi + ix);
            const float* wp = w + ((size_t)co * Cin * 9) + ky * 3 + kx;
            for (int ci = 0; ci < Cin; ++ci)
                acc += ip[(size_t)ci * HiWi] * wp[(size_t)ci * 9];
        }
    }
    if (relu) acc = fmaxf(acc, 0.0f);
    out[idx] = acc;
}

__global__ void bnstat_kernel(const float* __restrict__ x, float* __restrict__ st,
                              int Bn, int C, int HW) {
    int c = blockIdx.x;
    __shared__ float ss[256];
    __shared__ float sq[256];
    float s = 0, q = 0;
    for (int i = threadIdx.x; i < Bn * HW; i += blockDim.x) {
        int b = i / HW, r = i % HW;
        float v = x[((size_t)(b * C + c)) * HW + r];
        s += v;
        q += v * v;
    }
    ss[threadIdx.x] = s;
    sq[threadIdx.x] = q;
    __syncthreads();
    for (int st2 = 128; st2 > 0; st2 >>= 1) {
        if (threadIdx.x < st2) {
            ss[threadIdx.x] += ss[threadIdx.x + st2];
            sq[threadIdx.x] += sq[threadIdx.x + st2];
        }
        __syncthreads();
    }
    if (threadIdx.x == 0) {
        float n = (float)(Bn * HW);
        float mu = ss[0] / n;
        st[2 * c] = mu;
        st[2 * c + 1] = sq[0] / n - mu * mu;
    }
}

__global__ void bnrelu_kernel(const float* x, const float* st, const float* g,
                              const float* be, float* out, int C, int HW,
                              int total) {
    int idx = blockIdx.x * blockDim.x + threadIdx.x;
    if (idx >= total) return;
    int c = (idx / HW) % C;
    float mu = st[2 * c], var = st[2 * c + 1];
    out[idx] = fmaxf((x[idx] - mu) * rsqrtf(var + 1e-5f) * g[c] + be[c], 0.0f);
}

// convT with k == stride (non-overlapping); writes into concat buffer at choff
__global__ void convT_kernel(const float* in, const float* w, const float* bias,
                             float* out, int Bn, int C, int Hi, int Wi, int k,
                             int Cms, int choff) {
    int Ho = Hi * k, Wo = Wi * k;
    int idx = blockIdx.x * blockDim.x + threadIdx.x;
    int total = Bn * C * Ho * Wo;
    if (idx >= total) return;
    int x = idx % Wo;
    int y = (idx / Wo) % Ho;
    int co = (idx / (Wo * Ho)) % C;
    int b = idx / (Wo * Ho * C);
    int ky = y % k, kx = x % k, yi = y / k, xi = x / k;
    float acc = bias[co];
    for (int ci = 0; ci < C; ++ci)
        acc += in[(((size_t)(b * C + ci) * Hi) + yi) * Wi + xi] *
               w[(((size_t)(ci * C + co) * k) + ky) * k + kx];
    out[(((size_t)(b * Cms + choff + co) * Ho) + y) * Wo + x] = acc;
}

__global__ void copyf1_kernel(const float* f1, float* ms) {
    int idx = blockIdx.x * blockDim.x + threadIdx.x;
    if (idx >= 4 * 64 * 32 * 32) return;
    int x = idx & 31;
    int y = (idx >> 5) & 31;
    int c = (idx >> 10) & 63;
    int b = idx >> 16;
    ms[(((size_t)(b * 192 + c) * 32) + y) * 32 + x] = f1[idx];
}

__global__ void conv1x1_kernel(const float* in, const float* w, const float* bias,
                               float* out, int Bn, int Cin, int Cout, int HW) {
    int idx = blockIdx.x * blockDim.x + threadIdx.x;
    int total = Bn * Cout * HW;
    if (idx >= total) return;
    int r = idx % HW;
    int co = (idx / HW) % Cout;
    int b = idx / (HW * Cout);
    float acc = bias[co];
    for (int ci = 0; ci < Cin; ++ci)
        acc += in[((size_t)(b * Cin + ci)) * HW + r] * w[co * Cin + ci];
    out[idx] = acc;
}

__global__ void writeout_kernel(const float* y2, float* out) {
    int t = blockIdx.x * blockDim.x + threadIdx.x;
    if (t >= 4 * 32 * 32) return;
    int x = t & 31;
    int y = (t >> 5) & 31;
    int b = t >> 10;
    const float* src = y2 + ((size_t)b * 16) * 1024 + y * 32 + x;
    float* cls = out + (size_t)t * 10;
#pragma unroll
    for (int c = 0; c < 10; ++c) cls[c] = src[(size_t)c * 1024];
    float* reg = out + 40960 + (size_t)t * 7;
#pragma unroll
    for (int j = 0; j < 6; ++j) reg[j] = src[(size_t)(10 + j) * 1024];
    reg[6] = 0.0f;
}

// ---------------- host ----------------
static void conv_block(const float* in_, const float* wgt, const float* bias,
                       const float* g, const float* be, float* tmp, float* out_,
                       float* stats, int Bn, int C, int Hi, int stride,
                       hipStream_t s) {
    int Ho = Hi / stride;
    int total = Bn * C * Ho * Ho;
    conv3x3_kernel<<<(total + 255) / 256, 256, 0, s>>>(in_, wgt, bias, tmp, Bn, C,
                                                       C, Hi, Hi, Ho, Ho, stride, 0);
    bnstat_kernel<<<C, 256, 0, s>>>(tmp, stats, Bn, C, Ho * Ho);
    bnrelu_kernel<<<(total + 255) / 256, 256, 0, s>>>(tmp, stats, g, be, out_, C,
                                                      Ho * Ho, total);
}

extern "C" void kernel_launch(void* const* d_in, const int* in_sizes, int n_in,
                              void* d_out, int out_size, void* d_ws,
                              size_t ws_size, hipStream_t stream) {
    (void)in_sizes; (void)n_in; (void)out_size; (void)ws_size;
    const float* points = (const float*)d_in[0];
    const float* cc     = (const float*)d_in[1];
    const int*   mask   = (const int*)d_in[2];
#define F(i) ((const float*)d_in[i])

    char* wsp = (char*)d_ws;
    size_t off = 0;
    auto alloc = [&](size_t bytes) -> void* {
        void* p = wsp + off;
        off += (bytes + 255) & ~(size_t)255;
        return p;
    };
    float* bn     = (float*)alloc(256 * 4);
    float* stats  = (float*)alloc(512 * 4);
    float* pillar = (float*)alloc((size_t)8192 * 64 * 4);
    float* grid   = (float*)alloc((size_t)1048576 * 4);
    float* fn     = (float*)alloc((size_t)1048576 * 4);
    unsigned short* qb  = (unsigned short*)alloc((size_t)1048576 * 2);
    unsigned short* kbf = (unsigned short*)alloc((size_t)1048576 * 2);
    unsigned short* vtb = (unsigned short*)alloc((size_t)1048576 * 2);
    float* ob = (float*)alloc((size_t)1048576 * 4);
    float* tb = (float*)alloc((size_t)1048576 * 4);   // t, later reused as hh2
    float* hh = (float*)alloc((size_t)1048576 * 4);
    unsigned short* gb = (unsigned short*)alloc((size_t)16384 * 256 * 2);
    float* xb = (float*)alloc((size_t)1048576 * 4);
    unsigned short* wqT  = (unsigned short*)alloc(4096 * 2);
    unsigned short* wkT  = (unsigned short*)alloc(4096 * 2);
    unsigned short* wvT  = (unsigned short*)alloc(4096 * 2);
    unsigned short* woT  = (unsigned short*)alloc(4096 * 2);
    unsigned short* wf1T = (unsigned short*)alloc(16384 * 2);
    unsigned short* wf2T = (unsigned short*)alloc(16384 * 2);
    float* cbuf  = (float*)alloc((size_t)262144 * 4);
    float* pp    = (float*)alloc((size_t)262144 * 4);
    float* feat1 = (float*)alloc((size_t)262144 * 4);
    float* feat2 = (float*)alloc((size_t)65536 * 4);
    float* feat3 = (float*)alloc((size_t)16384 * 4);
    float* ms    = (float*)alloc((size_t)786432 * 4);
    float* yb    = (float*)alloc((size_t)786432 * 4);
    float* y2    = (float*)alloc((size_t)65536 * 4);

    // ---- init ----
    zero_kernel<<<1, 256, 0, stream>>>(bn, 256);
    zero_kernel<<<4096, 256, 0, stream>>>(grid, 1048576);
    prepw_kernel<<<64, 256, 0, stream>>>(F(11), F(13), F(15), F(17), F(23), F(25),
                                         wqT, wkT, wvT, woT, wf1T, wf2T);

    // ---- PFN ----
    pfn_pass1<<<1024, 256, 0, stream>>>(points, cc, F(3), F(4), bn);
    pfn_pass2<<<1024, 256, 0, stream>>>(points, cc, F(3), F(4), F(5), F(6), F(7),
                                        F(8), bn);
    pfn_pass3<<<1024, 256, 0, stream>>>(points, cc, F(3), F(4), F(5), F(6), F(7),
                                        F(8), F(9), F(10), bn, pillar);
    scatter_kernel<<<8192, 64, 0, stream>>>(pillar, cc, grid);

    // ---- attention block ----
    ln_kernel<<<2048, 256, 0, stream>>>(grid, F(19), F(20), fn, 16384);
    gemm_wmma_kernel<0><<<512, 256, 0, stream>>>(fn, wqT, F(12), 16384, 64, 64,
                                                 qb, 1, 0.125f, nullptr, 0);
    gemm_wmma_kernel<0><<<512, 256, 0, stream>>>(fn, wkT, F(14), 16384, 64, 64,
                                                 kbf, 1, 1.0f, nullptr, 0);
    gemm_wmma_kernel<0><<<512, 256, 0, stream>>>(fn, wvT, F(16), 16384, 64, 64,
                                                 vtb, 2, 1.0f, nullptr, 0);
    flash_attn<<<256, 128, 0, stream>>>(qb, kbf, vtb, mask, ob);
    gemm_wmma_kernel<0><<<512, 256, 0, stream>>>(ob, woT, F(18), 16384, 64, 64,
                                                 tb, 3, 1.0f, fn, 0);
    ln_kernel<<<2048, 256, 0, stream>>>(tb, F(21), F(22), hh, 16384);
    gemm_wmma_kernel<0><<<2048, 256, 0, stream>>>(hh, wf1T, F(24), 16384, 256, 64,
                                                  gb, 1, 1.0f, nullptr, 1);
    gemm_wmma_kernel<1><<<512, 256, 0, stream>>>(gb, wf2T, F(26), 16384, 64, 256,
                                                 tb, 3, 1.0f, hh, 0);
    buildx_kernel<<<4096, 256, 0, stream>>>(grid, tb, mask, xb);

    // ---- encoder stacks ----
    float* cur = xb;
    int Hc = 64;
    for (int i = 0; i < 3; ++i) {
        int st = (i == 0) ? 2 : 1;
        float* dst = (i == 2) ? feat1 : pp;
        conv_block(cur, F(27 + 4 * i), F(28 + 4 * i), F(29 + 4 * i), F(30 + 4 * i),
                   cbuf, dst, stats, 4, 64, Hc, st, stream);
        Hc /= st;
        cur = dst;
    }
    cur = feat1; Hc = 32;
    for (int i = 0; i < 5; ++i) {
        int st = (i == 0) ? 2 : 1;
        float* dst = (i == 4) ? feat2 : pp;
        conv_block(cur, F(39 + 4 * i), F(40 + 4 * i), F(41 + 4 * i), F(42 + 4 * i),
                   cbuf, dst, stats, 4, 64, Hc, st, stream);
        Hc /= st;
        cur = dst;
    }
    cur = feat2; Hc = 16;
    for (int i = 0; i < 5; ++i) {
        int st = (i == 0) ? 2 : 1;
        float* dst = (i == 4) ? feat3 : pp;
        conv_block(cur, F(59 + 4 * i), F(60 + 4 * i), F(61 + 4 * i), F(62 + 4 * i),
                   cbuf, dst, stats, 4, 64, Hc, st, stream);
        Hc /= st;
        cur = dst;
    }

    // ---- upsample + concat + heads ----
    convT_kernel<<<1024, 256, 0, stream>>>(feat2, F(79), F(80), ms, 4, 64, 16, 16,
                                           2, 192, 64);
    convT_kernel<<<1024, 256, 0, stream>>>(feat3, F(81), F(82), ms, 4, 64, 8, 8,
                                           4, 192, 128);
    copyf1_kernel<<<1024, 256, 0, stream>>>(feat1, ms);
    conv3x3_kernel<<<3072, 256, 0, stream>>>(ms, F(83), F(84), yb, 4, 192, 192,
                                             32, 32, 32, 32, 1, 1);
    conv1x1_kernel<<<256, 256, 0, stream>>>(yb, F(85), F(86), y2, 4, 192, 16, 1024);
    writeout_kernel<<<16, 256, 0, stream>>>(y2, (float*)d_out);
}
